// Decoder_76072460746940
// MI455X (gfx1250) — matrix operand order repaired and checked by reading, not compiled, not run on hardware
//
#include <hip/hip_runtime.h>
#include <cstddef>
#include <cstdint>

// ---------------------------------------------------------------------------
// Transformer decoder layer for MI455X (gfx1250, wave32, WMMA).
// All dense math runs through v_wmma_f32_16x16x32_bf16 (f32 accumulate).
// GEMM: 128x128 block tile, double-buffered LDS; next A-tile staged with
// CDNA5 GLOBAL_LOAD_ASYNC_TO_LDS_B128 (ASYNCcnt) overlapped with WMMAs.
// Attention is fused flash-style (S^T = K*Q^T so softmax is per-lane and the
// S^T C-layout doubles as the P A-fragment layout for the PV WMMA).
// Requires ws_size >= 176 MB.
// ---------------------------------------------------------------------------

typedef __bf16 bf16_t;
typedef __attribute__((ext_vector_type(16))) __bf16 v16bf;
typedef __attribute__((ext_vector_type(8)))  __bf16 v8bf;
typedef __attribute__((ext_vector_type(8)))  float   v8f;
typedef __attribute__((ext_vector_type(4)))  unsigned int u32x4;
typedef __attribute__((ext_vector_type(4)))  int          i32x4;

#define D_MODEL 1024
#define HEADS   16
#define DKH     64
#define BATCH   4
#define SEQ     1024
#define ROWS    (BATCH * SEQ)   // 4096
#define FFDIM   4096
#define NEGBIG  (-1e9f)

static __device__ __forceinline__ v8f vzero8() {
  v8f z = {0.f, 0.f, 0.f, 0.f, 0.f, 0.f, 0.f, 0.f};
  return z;
}

static __device__ __forceinline__ v16bf cat8(v8bf lo, v8bf hi) {
  v16bf r;
#pragma unroll
  for (int i = 0; i < 8; ++i) { r[i] = lo[i]; r[i + 8] = hi[i]; }
  return r;
}

// CDNA5 async global->LDS 16-byte copy (GVS mode: SGPR64 base + VGPR32 byte
// offset). Tracked by ASYNCcnt; bypasses VGPRs entirely.
static __device__ __forceinline__ void async_load_lds_b128(const void* gbase,
                                                           unsigned goff_bytes,
                                                           void* lds_ptr) {
  unsigned lds = (unsigned)(uintptr_t)lds_ptr;  // low 32 bits of generic = LDS addr
  asm volatile("global_load_async_to_lds_b128 %0, %1, %2"
               :: "v"(lds), "v"(goff_bytes), "s"(gbase)
               : "memory");
}
static __device__ __forceinline__ void wait_asynccnt0() {
  asm volatile("s_wait_asynccnt 0x0" ::: "memory");
}

// ---------------------------- f32 -> bf16 ----------------------------------
__global__ __launch_bounds__(256)
void cvt_bf16_kernel(const float* __restrict__ in, bf16_t* __restrict__ out, int n) {
  int i = blockIdx.x * 256 + threadIdx.x;
  if (i < n) out[i] = (bf16_t)in[i];
}

// ---------------------------- WMMA GEMM ------------------------------------
// C[M,N] = A[M,K] (bf16 row-major) * W[K,N] (bf16 row-major) + bias
// Block tile 128x128, 8 waves of 32x64 (2x4 WMMA 16x16 tiles), K-step 32.
// MODE: 0 = f32 row-major out, 1 = bf16 row-major out, 2 = bf16 transposed
//       per-batch out: out[b][col][row%rpb] (for attention V).
#define TM 128
#define TN 128
#define TK 32
#define LSTR 40   // padded LDS row stride in elems (80 B: 16-B aligned, no 4-way conflicts)

template <int MODE, bool RELU>
__global__ __launch_bounds__(256)
void gemm_bf16_kernel(const bf16_t* __restrict__ A, const bf16_t* __restrict__ W,
                      const float* __restrict__ bias,
                      float* __restrict__ outf, bf16_t* __restrict__ outb,
                      int M, int N, int K, int rpb) {
  __shared__ bf16_t sA[2][TM * LSTR];        // [m][k]
  __shared__ bf16_t sW[2][TN * LSTR];        // transposed: [n][k]
  const int tid  = threadIdx.x;
  const int wv   = tid >> 5;
  const int lane = tid & 31;
  const int l16  = lane & 15;
  const int hl   = lane >> 4;
  const int bm0  = blockIdx.y * TM;
  const int bn0  = blockIdx.x * TN;
  const int wm   = (wv >> 1) * 32;           // wave grid 4 (M) x 2 (N)
  const int wn   = (wv & 1) * 64;

  v8f acc[2][4];
#pragma unroll
  for (int mt = 0; mt < 2; ++mt)
#pragma unroll
    for (int nt = 0; nt < 4; ++nt) acc[mt][nt] = vzero8();

  // Stage the A tile (128x32) with async copies: 512 16-byte chunks, 2/thread.
  auto stage_A = [&](int k0, int buf) {
#pragma unroll
    for (int i = 0; i < 2; ++i) {
      int chunk = i * 256 + tid;
      int r = chunk >> 2;
      int c = (chunk & 3) * 8;
      unsigned goff = (unsigned)(((size_t)(bm0 + r) * K + (k0 + c)) * sizeof(bf16_t));
      async_load_lds_b128(A, goff, &sA[buf][r * LSTR + c]);
    }
  };
  // Stage the W tile (32x128) transposed into [n][k] (needs VGPR round-trip).
  auto stage_W = [&](int k0, int buf) {
#pragma unroll
    for (int i = 0; i < 2; ++i) {
      int chunk = i * 256 + tid;
      int kk = chunk >> 4;            // 0..31
      int nn = (chunk & 15) * 8;      // 0..120
      u32x4 d = *(const u32x4*)(W + (size_t)(k0 + kk) * N + bn0 + nn);
      const bf16_t* e = (const bf16_t*)&d;
#pragma unroll
      for (int j = 0; j < 8; ++j) sW[buf][(nn + j) * LSTR + kk] = e[j];
    }
  };

  int cur = 0;
  stage_A(0, 0);
  stage_W(0, 0);
  wait_asynccnt0();
  __syncthreads();

  for (int k0 = 0; k0 < K; k0 += TK) {
    // Kick off staging of the next tile into the other buffer, then compute.
    if (k0 + TK < K) {
      stage_A(k0 + TK, cur ^ 1);
      stage_W(k0 + TK, cur ^ 1);
    }
    const bf16_t* pA = sA[cur];
    const bf16_t* pW = sW[cur];

    // A-fragment (16x32, 16-bit): lanes 0-15 hold K 0-7 & 16-23; 16-31 hold 8-15 & 24-31.
    v16bf af[2], wf[4];
#pragma unroll
    for (int mt = 0; mt < 2; ++mt) {
      int row = wm + mt * 16 + l16;
      v8bf lo = *(const v8bf*)(pA + row * LSTR + hl * 8);
      v8bf hi = *(const v8bf*)(pA + row * LSTR + 16 + hl * 8);
      af[mt] = cat8(lo, hi);
    }
    // B-fragment (32x16): lanes 0-15 hold K 0-7 & 16-23; 16-31 hold 8-15 & 24-31.
#pragma unroll
    for (int nt = 0; nt < 4; ++nt) {
      int col = wn + nt * 16 + l16;
      v8bf lo = *(const v8bf*)(pW + col * LSTR + hl * 8);
      v8bf hi = *(const v8bf*)(pW + col * LSTR + 16 + hl * 8);
      wf[nt] = cat8(lo, hi);
    }
#pragma unroll
    for (int mt = 0; mt < 2; ++mt)
#pragma unroll
      for (int nt = 0; nt < 4; ++nt)
        acc[mt][nt] = __builtin_amdgcn_wmma_f32_16x16x32_bf16(
            false, af[mt], false, wf[nt], (short)0, acc[mt][nt], false, false);

    wait_asynccnt0();    // next-tile async copies landed in LDS
    __syncthreads();     // compiler adds DS waits; all waves flip together
    cur ^= 1;
  }

  // Epilogue. C layout: lane column = l16, rows = r + 8*hl.
#pragma unroll
  for (int mt = 0; mt < 2; ++mt) {
#pragma unroll
    for (int nt = 0; nt < 4; ++nt) {
      int col = bn0 + wn + nt * 16 + l16;
      float bv = bias[col];
      if (MODE == 2) {
        v8bf ov;
#pragma unroll
        for (int r = 0; r < 8; ++r) {
          float v = acc[mt][nt][r] + bv;
          if (RELU) v = fmaxf(v, 0.f);
          ov[r] = (bf16_t)v;
        }
        int row0 = bm0 + wm + mt * 16 + hl * 8;   // 8 consecutive rows -> 16B store
        int bi = row0 / rpb;
        int rl = row0 - bi * rpb;
        *(v8bf*)(outb + (size_t)bi * N * rpb + (size_t)col * rpb + rl) = ov;
      } else {
#pragma unroll
        for (int r = 0; r < 8; ++r) {
          int row = bm0 + wm + mt * 16 + hl * 8 + r;
          float v = acc[mt][nt][r] + bv;
          if (RELU) v = fmaxf(v, 0.f);
          if (MODE == 0) outf[(size_t)row * N + col] = v;
          else           outb[(size_t)row * N + col] = (bf16_t)v;
        }
      }
    }
  }
}

// ------------------------- Flash attention ---------------------------------
// One wave = one (b, h, 16-query tile). S^T = K * Q^T so the softmax is a
// per-lane reduction (+1 shfl_xor 16) and S^T's C layout is exactly the P
// A-fragment layout for the PV WMMA over a 32-key chunk.
// Q,K: bf16 row-major [B*SEQ, 1024] (col = h*64+d).  Vt: bf16 [b][h*64+d][key].
template <bool CAUSAL>
__global__ __launch_bounds__(256)
void attn_kernel(const bf16_t* __restrict__ Q, const bf16_t* __restrict__ Kx,
                 const bf16_t* __restrict__ Vt, const int* __restrict__ mask,
                 bf16_t* __restrict__ O) {
  const int tid  = threadIdx.x;
  const int wv   = tid >> 5;
  const int lane = tid & 31;
  const int l16  = lane & 15;
  const int hl   = lane >> 4;
  const int task = blockIdx.x * 8 + wv;      // B*H*(SEQ/16) = 4096 tasks
  const int qt = task & 63;
  const int h  = (task >> 6) & 15;
  const int b  = task >> 10;
  const int q0 = qt * 16;

  // Q^T B-fragments: lane n = query, elems = dk hl*8..+7 and 16+hl*8..+7 per step.
  const bf16_t* qrow = Q + (size_t)(b * SEQ + q0 + l16) * D_MODEL + h * DKH;
  v16bf qf0 = cat8(*(const v8bf*)(qrow + hl * 8),      *(const v8bf*)(qrow + 16 + hl * 8));
  v16bf qf1 = cat8(*(const v8bf*)(qrow + 32 + hl * 8), *(const v8bf*)(qrow + 48 + hl * 8));

  float m_run = -3.0e38f;
  float l_run = 0.f;
  v8f oacc[4];
#pragma unroll
  for (int dt = 0; dt < 4; ++dt) oacc[dt] = vzero8();

  const int nch = SEQ / 32;
  int jmax = nch - 1;
  if (CAUSAL) { int c = (q0 + 15) >> 5; if (c < jmax) jmax = c; }

  for (int jc = 0; jc <= jmax; ++jc) {
    v8f st[2];
#pragma unroll
    for (int j = 0; j < 2; ++j) {
      const int key0 = jc * 32 + j * 16;
      // K A-fragments (A = K tile, M = key, K-dim = dk).
      const bf16_t* krow = Kx + (size_t)(b * SEQ + key0 + l16) * D_MODEL + h * DKH;
      v16bf kf0 = cat8(*(const v8bf*)(krow + hl * 8),      *(const v8bf*)(krow + 16 + hl * 8));
      v16bf kf1 = cat8(*(const v8bf*)(krow + 32 + hl * 8), *(const v8bf*)(krow + 48 + hl * 8));
      v8f a = __builtin_amdgcn_wmma_f32_16x16x32_bf16(false, kf0, false, qf0, (short)0, vzero8(), false, false);
      a = __builtin_amdgcn_wmma_f32_16x16x32_bf16(false, kf1, false, qf1, (short)0, a, false, false);
      // This lane: query = q0+l16 (column), keys = key0 + hl*8 + r (rows).
      const int* mrow = mask + b * SEQ + key0 + hl * 8;
      i32x4 mv0 = *(const i32x4*)(mrow);
      i32x4 mv1 = *(const i32x4*)(mrow + 4);
      const int q = q0 + l16;
#pragma unroll
      for (int r = 0; r < 8; ++r) {
        int key = key0 + hl * 8 + r;
        int mv  = (r < 4) ? mv0[r] : mv1[r - 4];
        float s = a[r] * 0.125f;               // 1/sqrt(64)
        bool dead = (mv == 0) || (CAUSAL && key > q);
        a[r] = s + (dead ? NEGBIG : 0.f);
      }
      st[j] = a;
    }
    // Online softmax for this 32-key chunk (partner lane = lane ^ 16).
    float cm = -3.0e38f;
#pragma unroll
    for (int j = 0; j < 2; ++j)
#pragma unroll
      for (int r = 0; r < 8; ++r) cm = fmaxf(cm, st[j][r]);
    cm = fmaxf(cm, __shfl_xor(cm, 16, 32));
    float m_new = fmaxf(m_run, cm);
    float corr = __expf(m_run - m_new);
    float ps = 0.f;
#pragma unroll
    for (int j = 0; j < 2; ++j)
#pragma unroll
      for (int r = 0; r < 8; ++r) {
        float p = __expf(st[j][r] - m_new);
        st[j][r] = p;
        ps += p;
      }
    ps += __shfl_xor(ps, 16, 32);
    l_run = l_run * corr + ps;
    m_run = m_new;
    // Rescale O rows: O row query = hl*8 + r; its corr lives in lane (hl*8+r).
#pragma unroll
    for (int r = 0; r < 8; ++r) {
      float cr = __shfl(corr, hl * 8 + r, 32);
#pragma unroll
      for (int dt = 0; dt < 4; ++dt) oacc[dt][r] *= cr;
    }
    // P A-fragment: elems 0..7 = st[0] rows, 8..15 = st[1] rows (layout match).
    v16bf pf;
#pragma unroll
    for (int r = 0; r < 8; ++r) { pf[r] = (bf16_t)st[0][r]; pf[8 + r] = (bf16_t)st[1][r]; }
    // V B-fragments from transposed V, then PV WMMAs.
#pragma unroll
    for (int dt = 0; dt < 4; ++dt) {
      const bf16_t* vb = Vt + ((size_t)b * D_MODEL + h * DKH + dt * 16 + l16) * SEQ
                            + jc * 32 + hl * 8;
      v16bf vf = cat8(*(const v8bf*)(vb), *(const v8bf*)(vb + 16));
      oacc[dt] = __builtin_amdgcn_wmma_f32_16x16x32_bf16(
          false, pf, false, vf, (short)0, oacc[dt], false, false);
    }
  }
  // Finalize: divide each query row by its softmax sum, store bf16 row-major.
  float linv = 1.f / l_run;
#pragma unroll
  for (int r = 0; r < 8; ++r) {
    float li = __shfl(linv, hl * 8 + r, 32);
#pragma unroll
    for (int dt = 0; dt < 4; ++dt) {
      float v = oacc[dt][r] * li;
      O[(size_t)(b * SEQ + q0 + hl * 8 + r) * D_MODEL + h * DKH + dt * 16 + l16] = (bf16_t)v;
    }
  }
}

// ----------------------- residual + LayerNorm ------------------------------
__global__ __launch_bounds__(256)
void add_ln_kernel(const float* __restrict__ a, const float* __restrict__ y,
                   const float* __restrict__ g, const float* __restrict__ be,
                   float* __restrict__ outf, bf16_t* __restrict__ outb) {
  __shared__ float red[256];
  const int row = blockIdx.x;
  const int tid = threadIdx.x;
  const float* pa = a + (size_t)row * D_MODEL;
  const float* py = y + (size_t)row * D_MODEL;
  float v[4];
  float s = 0.f;
#pragma unroll
  for (int i = 0; i < 4; ++i) {
    int idx = tid + i * 256;
    float t = pa[idx] + py[idx];
    v[i] = t; s += t;
  }
  red[tid] = s; __syncthreads();
  for (int off = 128; off > 0; off >>= 1) {
    if (tid < off) red[tid] += red[tid + off];
    __syncthreads();
  }
  float mu = red[0] * (1.f / D_MODEL);
  __syncthreads();
  float s2 = 0.f;
#pragma unroll
  for (int i = 0; i < 4; ++i) { float d = v[i] - mu; s2 += d * d; }
  red[tid] = s2; __syncthreads();
  for (int off = 128; off > 0; off >>= 1) {
    if (tid < off) red[tid] += red[tid + off];
    __syncthreads();
  }
  float rstd = rsqrtf(red[0] * (1.f / D_MODEL) + 1e-6f);
#pragma unroll
  for (int i = 0; i < 4; ++i) {
    int idx = tid + i * 256;
    float o = (v[i] - mu) * rstd * g[idx] + be[idx];
    outf[(size_t)row * D_MODEL + idx] = o;
    if (outb) outb[(size_t)row * D_MODEL + idx] = (bf16_t)o;
  }
}

// ---------------------------------------------------------------------------
extern "C" void kernel_launch(void* const* d_in, const int* in_sizes, int n_in,
                              void* d_out, int out_size, void* d_ws, size_t ws_size,
                              hipStream_t stream) {
  (void)in_sizes; (void)n_in; (void)out_size; (void)ws_size;
  // Inputs: x, enc_y, enc_mask, dec_mask, then params flattened with sorted
  // dict keys: ca{k,o,q,v}{b,w}, ff1{b,w}, ff2{b,w}, ln1{b,g}, ln2{b,g},
  // ln3{b,g}, sa{k,o,q,v}{b,w}.
  const float* x     = (const float*)d_in[0];
  const float* ey    = (const float*)d_in[1];
  const int*   enc_m = (const int*)d_in[2];
  const int*   dec_m = (const int*)d_in[3];
  const float* ca_kb = (const float*)d_in[4];
  const float* ca_kw = (const float*)d_in[5];
  const float* ca_ob = (const float*)d_in[6];
  const float* ca_ow = (const float*)d_in[7];
  const float* ca_qb = (const float*)d_in[8];
  const float* ca_qw = (const float*)d_in[9];
  const float* ca_vb = (const float*)d_in[10];
  const float* ca_vw = (const float*)d_in[11];
  const float* ff1_b = (const float*)d_in[12];
  const float* ff1_w = (const float*)d_in[13];
  const float* ff2_b = (const float*)d_in[14];
  const float* ff2_w = (const float*)d_in[15];
  const float* ln1_b = (const float*)d_in[16];
  const float* ln1_g = (const float*)d_in[17];
  const float* ln2_b = (const float*)d_in[18];
  const float* ln2_g = (const float*)d_in[19];
  const float* ln3_b = (const float*)d_in[20];
  const float* ln3_g = (const float*)d_in[21];
  const float* sa_kb = (const float*)d_in[22];
  const float* sa_kw = (const float*)d_in[23];
  const float* sa_ob = (const float*)d_in[24];
  const float* sa_ow = (const float*)d_in[25];
  const float* sa_qb = (const float*)d_in[26];
  const float* sa_qw = (const float*)d_in[27];
  const float* sa_vb = (const float*)d_in[28];
  const float* sa_vw = (const float*)d_in[29];

  char* ws = (char*)d_ws;
  const size_t MB = 1024ull * 1024ull;
  bf16_t* w_saq = (bf16_t*)(ws + 0 * MB);
  bf16_t* w_sak = (bf16_t*)(ws + 2 * MB);
  bf16_t* w_sav = (bf16_t*)(ws + 4 * MB);
  bf16_t* w_sao = (bf16_t*)(ws + 6 * MB);
  bf16_t* w_caq = (bf16_t*)(ws + 8 * MB);
  bf16_t* w_cak = (bf16_t*)(ws + 10 * MB);
  bf16_t* w_cav = (bf16_t*)(ws + 12 * MB);
  bf16_t* w_cao = (bf16_t*)(ws + 14 * MB);
  bf16_t* w_ff1 = (bf16_t*)(ws + 16 * MB);   // 8 MB
  bf16_t* w_ff2 = (bf16_t*)(ws + 24 * MB);   // 8 MB
  bf16_t* xb    = (bf16_t*)(ws + 32 * MB);
  bf16_t* eb    = (bf16_t*)(ws + 40 * MB);
  bf16_t* qb    = (bf16_t*)(ws + 48 * MB);
  bf16_t* kb    = (bf16_t*)(ws + 56 * MB);
  bf16_t* vt    = (bf16_t*)(ws + 64 * MB);
  bf16_t* atb   = (bf16_t*)(ws + 72 * MB);
  float*  ybuf  = (float*)(ws + 80 * MB);    // 16 MB
  float*  x1f   = (float*)(ws + 96 * MB);    // 16 MB
  bf16_t* x1b   = (bf16_t*)(ws + 112 * MB);
  float*  x2f   = (float*)(ws + 120 * MB);   // 16 MB
  bf16_t* x2b   = (bf16_t*)(ws + 136 * MB);
  bf16_t* hb    = (bf16_t*)(ws + 144 * MB);  // 32 MB -> 176 MB total

  auto cvt = [&](const float* src, bf16_t* dst, int n) {
    cvt_bf16_kernel<<<(n + 255) / 256, 256, 0, stream>>>(src, dst, n);
  };
  cvt(x, xb, ROWS * D_MODEL);
  cvt(ey, eb, ROWS * D_MODEL);
  cvt(sa_qw, w_saq, D_MODEL * D_MODEL);
  cvt(sa_kw, w_sak, D_MODEL * D_MODEL);
  cvt(sa_vw, w_sav, D_MODEL * D_MODEL);
  cvt(sa_ow, w_sao, D_MODEL * D_MODEL);
  cvt(ca_qw, w_caq, D_MODEL * D_MODEL);
  cvt(ca_kw, w_cak, D_MODEL * D_MODEL);
  cvt(ca_vw, w_cav, D_MODEL * D_MODEL);
  cvt(ca_ow, w_cao, D_MODEL * D_MODEL);
  cvt(ff1_w, w_ff1, D_MODEL * FFDIM);
  cvt(ff2_w, w_ff2, FFDIM * D_MODEL);

  dim3 gDD(D_MODEL / TN, ROWS / TM);   // (8, 32)
  dim3 gDF(FFDIM / TN, ROWS / TM);     // (32, 32)

  // --- self-attention block ---
  gemm_bf16_kernel<1, false><<<gDD, 256, 0, stream>>>(xb, w_saq, sa_qb, nullptr, qb, ROWS, D_MODEL, D_MODEL, 0);
  gemm_bf16_kernel<1, false><<<gDD, 256, 0, stream>>>(xb, w_sak, sa_kb, nullptr, kb, ROWS, D_MODEL, D_MODEL, 0);
  gemm_bf16_kernel<2, false><<<gDD, 256, 0, stream>>>(xb, w_sav, sa_vb, nullptr, vt, ROWS, D_MODEL, D_MODEL, SEQ);
  attn_kernel<true><<<512, 256, 0, stream>>>(qb, kb, vt, dec_m, atb);
  gemm_bf16_kernel<0, false><<<gDD, 256, 0, stream>>>(atb, w_sao, sa_ob, ybuf, nullptr, ROWS, D_MODEL, D_MODEL, 0);
  add_ln_kernel<<<ROWS, 256, 0, stream>>>(x, ybuf, ln1_g, ln1_b, x1f, x1b);

  // --- cross-attention block ---
  gemm_bf16_kernel<1, false><<<gDD, 256, 0, stream>>>(x1b, w_caq, ca_qb, nullptr, qb, ROWS, D_MODEL, D_MODEL, 0);
  gemm_bf16_kernel<1, false><<<gDD, 256, 0, stream>>>(eb,  w_cak, ca_kb, nullptr, kb, ROWS, D_MODEL, D_MODEL, 0);
  gemm_bf16_kernel<2, false><<<gDD, 256, 0, stream>>>(eb,  w_cav, ca_vb, nullptr, vt, ROWS, D_MODEL, D_MODEL, SEQ);
  attn_kernel<false><<<512, 256, 0, stream>>>(qb, kb, vt, enc_m, atb);
  gemm_bf16_kernel<0, false><<<gDD, 256, 0, stream>>>(atb, w_cao, ca_ob, ybuf, nullptr, ROWS, D_MODEL, D_MODEL, 0);
  add_ln_kernel<<<ROWS, 256, 0, stream>>>(x1f, ybuf, ln2_g, ln2_b, x2f, x2b);

  // --- FFN block ---
  gemm_bf16_kernel<1, true><<<gDF, 256, 0, stream>>>(x2b, w_ff1, ff1_b, nullptr, hb, ROWS, FFDIM, D_MODEL, 0);
  gemm_bf16_kernel<0, false><<<gDD, 256, 0, stream>>>(hb, w_ff2, ff2_b, ybuf, nullptr, ROWS, D_MODEL, FFDIM, 0);
  add_ln_kernel<<<ROWS, 256, 0, stream>>>(x2f, ybuf, ln3_g, ln3_b, (float*)d_out, nullptr);
}
